// HGT_19748259627257
// MI455X (gfx1250) — compile-verified, matrix operand order checked by reading, not run on hardware
//
#include <hip/hip_runtime.h>
#include <math.h>

#define NN1 10000
#define NN2 10000
#define FHID 128
#define NH 8
#define HD 16
#define NLAYER 2
#define WT_STRIDE 136   // padded LDS row stride (bank-conflict-free B reads)

typedef __bf16 bf16_t;
typedef bf16_t v16bf __attribute__((ext_vector_type(16)));
typedef float  v8f   __attribute__((ext_vector_type(8)));

enum { PRE_NONE = 0, PRE_GELU = 1 };
enum { POST_NONE = 0, POST_RELU = 1, POST_SKIP = 2 };
enum { W_FULL = 0, W_BLOCKDIAG = 1 };

__device__ __forceinline__ float gelu_exact(float x) {
  return 0.5f * x * (1.0f + erff(x * 0.70710678118654752f));
}

// ---------------------------------------------------------------------------
// Fused 128-wide GEMM:  Y = post( A_pre(X) @ W + b )   with WMMA bf16->f32.
// Block = 128 threads = 4 waves; each wave computes a 16x128 output strip.
// Weight (f32, [128 k][128 n] row-major, or [H,16,16] block-diagonal) is
// converted to bf16 and transposed into LDS as Wt[n][k] once per block.
// ---------------------------------------------------------------------------
__global__ __launch_bounds__(128)
void gemm128_wmma(const float* __restrict__ X, const float* __restrict__ W,
                  const float* __restrict__ bias, const float* __restrict__ Xold,
                  const float* __restrict__ skip_logit,
                  float* __restrict__ Y, float* __restrict__ Y2, int ldy2,
                  int nrows, int pre, int post, int wmode)
{
  __shared__ bf16_t Wt[FHID * WT_STRIDE];
  __shared__ float  bs[FHID];
  const int tid = threadIdx.x;

  if (wmode == W_FULL) {
    for (int i = tid; i < FHID * FHID; i += 128) {
      int k = i >> 7, n = i & 127;
      Wt[n * WT_STRIDE + k] = (bf16_t)W[i];
    }
  } else {
    for (int i = tid; i < FHID * WT_STRIDE; i += 128) Wt[i] = (bf16_t)0.0f;
    __syncthreads();
    for (int i = tid; i < NH * HD * HD; i += 128) {
      int h = i >> 8, d = (i >> 4) & 15, e = i & 15;
      // W_big[k = h*16+d][n = h*16+e] = a_rel[h][d][e]
      Wt[(h * 16 + e) * WT_STRIDE + (h * 16 + d)] = (bf16_t)W[i];
    }
  }
  bs[tid] = bias ? bias[tid] : 0.0f;
  __syncthreads();

  const int wave  = tid >> 5;
  const int lane  = tid & 31;
  const int lhalf = lane >> 4;   // 0 or 1
  const int l15   = lane & 15;
  const int kh    = lhalf * 16;  // K half offset: contiguous k = 16*lhalf + j, same for A and B

  float sblend = 0.0f;
  if (post == POST_SKIP) sblend = 1.0f / (1.0f + __expf(-skip_logit[0]));

  const int rbase = blockIdx.x * 64 + wave * 16;
  if (rbase >= nrows) return;  // wave-uniform exit; no barriers after this

  // A fragments: lane<16 holds row M=l15 K 0..15; lane>=16 same row K 16..31.
  const int row  = rbase + l15;
  const int rowc = row < nrows ? row : nrows - 1;
  v16bf afrag[4];
#pragma unroll
  for (int c = 0; c < 4; ++c) {
    const float* sp = X + (size_t)rowc * FHID + c * 32 + kh;
#pragma unroll
    for (int j = 0; j < 16; ++j) {
      float v = sp[j];
      if (pre == PRE_GELU) v = gelu_exact(v);
      afrag[c][j] = (bf16_t)v;
    }
  }

#pragma unroll
  for (int t = 0; t < 8; ++t) {
    v8f acc = {};
    const int n = t * 16 + l15;
#pragma unroll
    for (int c = 0; c < 4; ++c) {
      v16bf bfrag;
      const bf16_t* wp = &Wt[n * WT_STRIDE + c * 32 + kh];
#pragma unroll
      for (int j = 0; j < 16; ++j) bfrag[j] = wp[j];
      acc = __builtin_amdgcn_wmma_f32_16x16x32_bf16(
          false, afrag[c], false, bfrag, (short)0, acc, false, false);
    }
    // C/D layout: VGPR r -> M = r (lanes 0-15) or 8+r (lanes 16-31), N = lane&15
    const float bval = bs[n];
    const int   mb   = lhalf * 8;
#pragma unroll
    for (int r = 0; r < 8; ++r) {
      int orow = rbase + mb + r;
      if (orow < nrows) {
        float o = acc[r] + bval;
        if (post == POST_RELU)      o = fmaxf(o, 0.0f);
        else if (post == POST_SKIP) o = sblend * o + (1.0f - sblend) * Xold[(size_t)orow * FHID + n];
        Y[(size_t)orow * FHID + n] = o;
        if (Y2) Y2[(size_t)orow * ldy2 + n] = o;
      }
    }
  }
}

// ---------------------------------------------------------------------------
// Edge-attention kernels (L2-resident gathers + atomics)
// ---------------------------------------------------------------------------
__device__ __forceinline__ unsigned ordf(float x) {
  unsigned b = __float_as_uint(x);
  return (b & 0x80000000u) ? ~b : (b | 0x80000000u);
}
__device__ __forceinline__ float unordf(unsigned u) {
  unsigned b = (u & 0x80000000u) ? (u & 0x7FFFFFFFu) : ~u;
  return __uint_as_float(b);
}

__global__ void fill_u32(unsigned* __restrict__ p, unsigned v, int n) {
  int i = blockIdx.x * blockDim.x + threadIdx.x;
  if (i < n) p[i] = v;
}

__global__ void edge_alpha_max(const int* __restrict__ src, const int* __restrict__ dst,
                               const float* __restrict__ q, const float* __restrict__ ke,
                               const float* __restrict__ p_rel,
                               float* __restrict__ alpha, unsigned* __restrict__ segmax, int E)
{
  int idx = blockIdx.x * blockDim.x + threadIdx.x;
  if (idx >= E * NH) return;
  int e = idx >> 3, h = idx & 7;
  int s = src[e], d = dst[e];
  const float* qp = q  + (size_t)d * FHID + h * HD;
  const float* kp = ke + (size_t)s * FHID + h * HD;
  float acc = 0.0f;
#pragma unroll
  for (int j = 0; j < HD; ++j) acc = fmaf(qp[j], kp[j], acc);
  float a = acc * p_rel[h] * 0.25f;  // 1/sqrt(D), D=16
  alpha[idx] = a;
  atomicMax(&segmax[d * NH + h], ordf(a));
}

__global__ void edge_expsum(const int* __restrict__ dst, float* __restrict__ alpha,
                            const unsigned* __restrict__ segmax,
                            float* __restrict__ segsum, int E)
{
  int idx = blockIdx.x * blockDim.x + threadIdx.x;
  if (idx >= E * NH) return;
  int e = idx >> 3, h = idx & 7;
  int d = dst[e];
  float m  = unordf(segmax[d * NH + h]);
  float ev = __expf(alpha[idx] - m);
  alpha[idx] = ev;
  atomicAdd(&segsum[d * NH + h], ev);
}

__global__ void edge_aggregate(const int* __restrict__ src, const int* __restrict__ dst,
                               const float* __restrict__ alpha, const float* __restrict__ segsum,
                               const float* __restrict__ ve, float* __restrict__ agg, int total)
{
  int idx = blockIdx.x * blockDim.x + threadIdx.x;
  if (idx >= total) return;
  int e = idx >> 7;
  int c = idx & 127;
  int h = c >> 4;
  int s = src[e], d = dst[e];
  float w = alpha[(size_t)e * NH + h] / (segsum[d * NH + h] + 1e-16f);
  atomicAdd(&agg[(size_t)d * FHID + c], w * ve[(size_t)s * FHID + c]);
}

// One wave per query edge: 256-dim gather-dot with shfl reduction.
__global__ __launch_bounds__(256)
void final_gather_dot(const int* __restrict__ mi, const int* __restrict__ di,
                      const float* __restrict__ Em, const float* __restrict__ Ed,
                      float* __restrict__ y, int EF)
{
  int gid  = blockIdx.x * blockDim.x + threadIdx.x;
  int wid  = gid >> 5;
  int lane = threadIdx.x & 31;
  if (wid >= EF) return;
  const float* a = Em + (size_t)mi[wid] * 256;
  const float* b = Ed + (size_t)di[wid] * 256;
  float acc = 0.0f;
#pragma unroll
  for (int j = 0; j < 8; ++j) acc = fmaf(a[lane + 32 * j], b[lane + 32 * j], acc);
#pragma unroll
  for (int off = 16; off > 0; off >>= 1) acc += __shfl_xor(acc, off, 32);
  if (lane == 0) y[wid] = acc;
}

// ---------------------------------------------------------------------------
extern "C" void kernel_launch(void* const* d_in, const int* in_sizes, int n_in,
                              void* d_out, int out_size, void* d_ws, size_t ws_size,
                              hipStream_t stream)
{
  (void)n_in; (void)out_size; (void)ws_size;

  const float* x_n1  = (const float*)d_in[0];
  const float* x_n2  = (const float*)d_in[1];
  const float* W_in1 = (const float*)d_in[2];
  const float* b_in1 = (const float*)d_in[3];
  const float* W_in2 = (const float*)d_in[4];
  const float* b_in2 = (const float*)d_in[5];
  const float* Wk_t[2]   = {(const float*)d_in[6],  (const float*)d_in[15]};
  const float* bk_t[2]   = {(const float*)d_in[7],  (const float*)d_in[16]};
  const float* Wq_t[2]   = {(const float*)d_in[8],  (const float*)d_in[17]};
  const float* bq_t[2]   = {(const float*)d_in[9],  (const float*)d_in[18]};
  const float* Wv_t[2]   = {(const float*)d_in[10], (const float*)d_in[19]};
  const float* bv_t[2]   = {(const float*)d_in[11], (const float*)d_in[20]};
  const float* Wa_t[2]   = {(const float*)d_in[12], (const float*)d_in[21]};
  const float* ba_t[2]   = {(const float*)d_in[13], (const float*)d_in[22]};
  const float* skip_t[2] = {(const float*)d_in[14], (const float*)d_in[23]};
  const float* a_rel[2]  = {(const float*)d_in[24], (const float*)d_in[27]}; // rel 12, 21
  const float* m_rel[2]  = {(const float*)d_in[25], (const float*)d_in[28]};
  const float* p_rel[2]  = {(const float*)d_in[26], (const float*)d_in[29]};
  const int* ei12 = (const int*)d_in[30];
  const int* ei21 = (const int*)d_in[31];
  const int* eidx = (const int*)d_in[32];
  float* y = (float*)d_out;

  const int E12 = in_sizes[30] / 2;
  const int E21 = in_sizes[31] / 2;
  const int EF  = in_sizes[32] / 2;
  const int Emax = E12 > E21 ? E12 : E21;

  // ---- workspace layout (f32 elements) ----
  float* ws = (float*)d_ws;
  const size_t NODE = (size_t)NN1 * FHID;
  float* x[2];     x[0] = ws; ws += NODE; x[1] = ws; ws += NODE;
  float* kb[2];    float* qb[2];  float* vb[2];
  kb[0] = ws; ws += NODE; qb[0] = ws; ws += NODE; vb[0] = ws; ws += NODE;
  kb[1] = ws; ws += NODE; qb[1] = ws; ws += NODE; vb[1] = ws; ws += NODE;
  float* keb   = ws; ws += NODE;
  float* veb   = ws; ws += NODE;
  float* alpha = ws; ws += (size_t)Emax * NH;
  unsigned* segmax = (unsigned*)ws; ws += (size_t)NN1 * NH;
  float* segsum = ws; ws += (size_t)NN1 * NH;
  float* agg    = ws; ws += NODE;
  float* Em     = ws; ws += (size_t)NN1 * 2 * FHID;
  float* Ed     = ws; ws += (size_t)NN2 * 2 * FHID;

  auto gemm = [&](const float* X, const float* W, const float* bias,
                  const float* Xold, const float* skipl,
                  float* Y, float* Y2, int ldy2, int n, int pre, int post, int wmode) {
    gemm128_wmma<<<(n + 63) / 64, 128, 0, stream>>>(X, W, bias, Xold, skipl,
                                                    Y, Y2, ldy2, n, pre, post, wmode);
  };

  // input projections + relu
  gemm(x_n1, W_in1, b_in1, nullptr, nullptr, x[0], nullptr, 0, NN1, PRE_NONE, POST_RELU, W_FULL);
  gemm(x_n2, W_in2, b_in2, nullptr, nullptr, x[1], nullptr, 0, NN2, PRE_NONE, POST_RELU, W_FULL);

  for (int l = 0; l < NLAYER; ++l) {
    const size_t wOff = (size_t)l * FHID * FHID;
    const size_t bOff = (size_t)l * FHID;
    for (int t = 0; t < 2; ++t) {
      const int n = (t == 0) ? NN1 : NN2;
      gemm(x[t], Wk_t[t] + wOff, bk_t[t] + bOff, nullptr, nullptr, kb[t], nullptr, 0, n, PRE_NONE, POST_NONE, W_FULL);
      gemm(x[t], Wq_t[t] + wOff, bq_t[t] + bOff, nullptr, nullptr, qb[t], nullptr, 0, n, PRE_NONE, POST_NONE, W_FULL);
      gemm(x[t], Wv_t[t] + wOff, bv_t[t] + bOff, nullptr, nullptr, vb[t], nullptr, 0, n, PRE_NONE, POST_NONE, W_FULL);
    }

    for (int r = 0; r < 2; ++r) {            // r=0: n1->n2 ; r=1: n2->n1
      const int st = (r == 0) ? 0 : 1;
      const int dt = 1 - st;
      const int* ei = (r == 0) ? ei12 : ei21;
      const int  E  = (r == 0) ? E12 : E21;
      const int  ns = (st == 0) ? NN1 : NN2;
      const int  nd = (dt == 0) ? NN1 : NN2;
      const size_t relOff = (size_t)l * NH * HD * HD;

      // relation-specific K/V transforms as block-diagonal WMMA GEMMs
      gemm(kb[st], a_rel[r] + relOff, nullptr, nullptr, nullptr, keb, nullptr, 0, ns, PRE_NONE, POST_NONE, W_BLOCKDIAG);
      gemm(vb[st], m_rel[r] + relOff, nullptr, nullptr, nullptr, veb, nullptr, 0, ns, PRE_NONE, POST_NONE, W_BLOCKDIAG);

      const int segN = nd * NH;
      fill_u32<<<(segN + 255) / 256, 256, 0, stream>>>(segmax, 0x007FFFFFu, segN);  // ordered(-inf)
      fill_u32<<<(segN + 255) / 256, 256, 0, stream>>>((unsigned*)segsum, 0u, segN);
      fill_u32<<<((int)NODE + 255) / 256, 256, 0, stream>>>((unsigned*)agg, 0u, (int)NODE);

      const int* srcp = ei;
      const int* dstp = ei + E;
      const int eh = E * NH;
      edge_alpha_max<<<(eh + 255) / 256, 256, 0, stream>>>(srcp, dstp, qb[dt], keb,
                                                           p_rel[r] + (size_t)l * NH, alpha, segmax, E);
      edge_expsum<<<(eh + 255) / 256, 256, 0, stream>>>(dstp, alpha, segmax, segsum, E);
      const int tot = E * FHID;
      edge_aggregate<<<(tot + 255) / 256, 256, 0, stream>>>(srcp, dstp, alpha, segsum, veb, agg, tot);

      // out proj: x[dt] = s*(gelu(agg)@Wa + ba) + (1-s)*x[dt]; also write Em/Ed slice
      float* Eout = (dt == 0) ? Em : Ed;
      gemm(agg, Wa_t[dt] + wOff, ba_t[dt] + bOff, x[dt], skip_t[dt] + l,
           x[dt], Eout + (size_t)l * FHID, 2 * FHID, nd, PRE_GELU, POST_SKIP, W_FULL);
    }
  }

  final_gather_dot<<<(EF * 32 + 255) / 256, 256, 0, stream>>>(eidx, eidx + EF, Em, Ed, y, EF);
}
